// FeedbackAttention_46531675685296
// MI455X (gfx1250) — matrix operand order repaired and checked by reading, not compiled, not run on hardware
//
#include <hip/hip_runtime.h>
#include <hip/hip_bf16.h>
#include <cstdint>

// ---------------------------------------------------------------------------
// FeedbackAttention for MI455X (gfx1250), wave32.
//   - v_wmma_f32_16x16x32_f16 for all GEMMs (f32 accumulate, f16 operands)
//   - global_load_async_to_lds_b128 + s_wait_asynccnt for the `value` stream
//   - all B-operands pre-swizzled into WMMA fragment order (contiguous v16h
//     loads, no strided-scalar/wait serialization)
//   - query folded through w_k / w_v => ~9 GFLOP total, HBM-bound (~1.07 GB)
// Sizes hardcoded: HEADS=16, D_MODEL=1024, D_K=64, SEQ=2048, BATCH=64, P=4096.
// ---------------------------------------------------------------------------

typedef __attribute__((ext_vector_type(16))) _Float16 v16h;
typedef __attribute__((ext_vector_type(8)))  _Float16 v8h;
typedef __attribute__((ext_vector_type(8)))  float    v8f;

#define HEADS   16
#define DMODEL  1024
#define DK      64
#define SEQ     2048
#define BATCH   64
#define PPOS    4096
#define POFF    (PPOS - SEQ)   // 2048

// ---- WMMA fragment helpers (CDNA5 ISA 7.12.2 layouts, wave32) --------------
// A 16x32 f16: lane r=lane&15 holds row r; sel=lane>>4;
//   e=0..15 -> k=16*sel+e (contiguous; same k order as B)
// B 32x16 f16: lane n=lane&15 holds col n; sel=lane>>4; k=16*sel+e
// C/D f32 16x16: row = i+8*sel, col = lane&15

__device__ __forceinline__ v16h pack16(v8h lo, v8h hi) {
    return __builtin_shufflevector(lo, hi, 0, 1, 2, 3, 4, 5, 6, 7,
                                   8, 9, 10, 11, 12, 13, 14, 15);
}

__device__ __forceinline__ v8h cvt8(float4 x, float4 y) {
    v8h r;
    r[0] = (_Float16)x.x; r[1] = (_Float16)x.y; r[2] = (_Float16)x.z; r[3] = (_Float16)x.w;
    r[4] = (_Float16)y.x; r[5] = (_Float16)y.y; r[6] = (_Float16)y.z; r[7] = (_Float16)y.w;
    return r;
}

// A-fragment from row-major f32 (vectorized b128 loads; all call sites 16B aligned)
__device__ __forceinline__ v16h frag_a_f32(const float* base, int ld, int lane) {
    const int r = lane & 15, sel = lane >> 4;
    const float* p = base + (size_t)r * ld + 16 * sel;
    const float4* q0 = (const float4*)p;
    const float4* q1 = (const float4*)(p + 8);
    return pack16(cvt8(q0[0], q0[1]), cvt8(q1[0], q1[1]));
}

// A-fragment from row-major f16
__device__ __forceinline__ v16h frag_a_f16(const _Float16* base, int ld, int lane) {
    const int r = lane & 15, sel = lane >> 4;
    const _Float16* p = base + (size_t)r * ld + 16 * sel;
    return pack16(*(const v8h*)p, *(const v8h*)(p + 8));
}

// B-fragment from transposed storage: element (k,n) = base[n*ld + k] (contiguous)
__device__ __forceinline__ v16h frag_bT_f32(const float* base, int ld, int lane) {
    const int n = lane & 15, sel = lane >> 4;
    const float4* p = (const float4*)(base + (size_t)n * ld + 16 * sel);
    return pack16(cvt8(p[0], p[1]), cvt8(p[2], p[3]));
}

__device__ __forceinline__ v8f wmma_f16(v16h a, v16h b, v8f c) {
    return __builtin_amdgcn_wmma_f32_16x16x32_f16(false, a, false, b, (short)0, c,
                                                  false, false);
}

// swizzled-B index: fragment tile (kk = k/32, nt = n/16), per-lane 16 halfs
__device__ __forceinline__ const v16h* bsw_tile(const _Float16* base, int kk, int nt,
                                                int lane) {
    return (const v16h*)(base + (((size_t)kk * (DMODEL / 16) + nt) * 32 + lane) * 16);
}

// ---------------------------------------------------------------------------
// K0: repack a 1024x1024 f32 weight matrix into B-fragment-ordered f16.
// Coalesced float4 reads, scattered (fire-and-forget) f16 writes.
// grid (32 k-chunks, 16 n-blocks), block 256
__global__ void krepack(const float* __restrict__ w, _Float16* __restrict__ wsw) {
    const int kk = blockIdx.x;            // 32-row k-chunk
    const int nb = blockIdx.y;            // 64-col block
    const int t = threadIdx.x;            // 0..255
#pragma unroll
    for (int p = 0; p < 2; ++p) {
        const int linear = p * 256 + t;   // 0..511
        const int row = linear >> 4;      // 0..31 within chunk
        const int seg = linear & 15;      // 16B segment within 64 cols
        const int k = kk * 32 + row;
        const float4 v4 = *(const float4*)(w + (size_t)k * DMODEL + nb * 64 + seg * 4);
        const int laneBase = (row >> 4) << 4;   // sel bit of target lane
        const int e = k & 15;
        const float vals[4] = {v4.x, v4.y, v4.z, v4.w};
#pragma unroll
        for (int u = 0; u < 4; ++u) {
            const int n_g = nb * 64 + seg * 4 + u;
            const int nt = n_g >> 4;
            const int L = laneBase | (n_g & 15);
            wsw[(((size_t)kk * (DMODEL / 16) + nt) * 32 + L) * 16 + e] = (_Float16)vals[u];
        }
    }
}

// ---------------------------------------------------------------------------
// K1: q = query @ w_q  (64x1024 @ 1024x1024) -> f32   (B pre-swizzled)
__global__ void k1_qproj(const float* __restrict__ query,
                         const _Float16* __restrict__ wq_sw,
                         float* __restrict__ q_out) {
    const int lane = threadIdx.x & 31;
    const int nt = blockIdx.x, b0 = blockIdx.y * 16;

    v16h a = frag_a_f32(query + (size_t)b0 * DMODEL, DMODEL, lane);
    v16h b = *bsw_tile(wq_sw, 0, nt, lane);
    v8f acc = {};
#pragma unroll 4
    for (int kk = 0; kk < DMODEL / 32 - 1; ++kk) {
        v16h a2 = frag_a_f32(query + (size_t)b0 * DMODEL + (kk + 1) * 32, DMODEL, lane);
        v16h b2 = *bsw_tile(wq_sw, kk + 1, nt, lane);
        acc = wmma_f16(a, b, acc);
        a = a2; b = b2;
    }
    acc = wmma_f16(a, b, acc);

    const int n = lane & 15, sel = lane >> 4;
#pragma unroll
    for (int i = 0; i < 8; ++i)
        q_out[(size_t)(b0 + i + 8 * sel) * DMODEL + nt * 16 + n] = acc[i];
}

// ---------------------------------------------------------------------------
// K2a: Afold[b][c][h] = sum_d (q[b,h,d]+qpb[h,d]) * w_k[c, h*64+d]
// stored PRE-SWIZZLED in K3's B-fragment order:
//   afold_sw[((b*32 + kk)*32 + lane)*16 + e],  k=c within chunk = 16*sel+e, n=h
__global__ void k2a_afold(const float* __restrict__ q,
                          const float* __restrict__ w_k,
                          const float* __restrict__ qpb,   // (16,64)
                          _Float16* __restrict__ afold_sw) {
    const int lane = threadIdx.x & 31;
    const int c0 = blockIdx.x * 16, b0 = blockIdx.y * 16, h = blockIdx.z;
    const int nb = lane & 15, sel = lane >> 4;
    v8f acc = {};
#pragma unroll
    for (int kk = 0; kk < DK / 32; ++kk) {
        const int d0 = kk * 32;
        v16h a = frag_a_f32(w_k + (size_t)c0 * DMODEL + h * DK + d0, DMODEL, lane);
        // B[k=d, n=b] = q[(b0+n)*1024 + h*64 + d0+k] + qpb[h*64 + d0+k]
        const float4* qp = (const float4*)(q + (size_t)(b0 + nb) * DMODEL + h * DK + d0 + 16 * sel);
        const float4* bp = (const float4*)(qpb + h * DK + d0 + 16 * sel);
        float4 s0 = qp[0], s1 = qp[1], s2 = qp[2], s3 = qp[3];
        float4 c0v = bp[0], c1v = bp[1], c2v = bp[2], c3v = bp[3];
        s0.x += c0v.x; s0.y += c0v.y; s0.z += c0v.z; s0.w += c0v.w;
        s1.x += c1v.x; s1.y += c1v.y; s1.z += c1v.z; s1.w += c1v.w;
        s2.x += c2v.x; s2.y += c2v.y; s2.z += c2v.z; s2.w += c2v.w;
        s3.x += c3v.x; s3.y += c3v.y; s3.z += c3v.z; s3.w += c3v.w;
        v16h b = pack16(cvt8(s0, s1), cvt8(s2, s3));
        acc = wmma_f16(a, b, acc);
    }
#pragma unroll
    for (int i = 0; i < 8; ++i) {
        const int c = c0 + i + 8 * sel;      // output row (c), col = nb (b), head h
        const int kk = c >> 5, cm = c & 31;
        const int selt = cm >> 4, et = cm & 15;
        const int L = (selt << 4) | h;
        afold_sw[(((size_t)(b0 + nb) * 32 + kk) * 32 + L) * 16 + et] = (_Float16)acc[i];
    }
}

// ---------------------------------------------------------------------------
// K2b: scores[b][h][j] = q[b,h,:] . kpe[j,:] + kpb[j,h]   (bd term)
__global__ void k2b_bd(const float* __restrict__ q,
                       const float* __restrict__ kpe,   // (4096,64)
                       const float* __restrict__ kpb,   // (4096,16)
                       float* __restrict__ scores) {
    const int lane = threadIdx.x & 31;
    const int j0 = blockIdx.x * 16, bt = blockIdx.y;
    v8f acc = {};
#pragma unroll
    for (int kk = 0; kk < DK / 32; ++kk) {
        const int d0 = kk * 32;
        v16h a = frag_a_f32(q + (size_t)bt * DMODEL + d0, DK, lane);
        v16h b = frag_bT_f32(kpe + (size_t)(POFF + j0) * DK + d0, DK, lane);
        acc = wmma_f16(a, b, acc);
    }
    const int n = lane & 15, sel = lane >> 4;
    const int j = j0 + n;
#pragma unroll
    for (int i = 0; i < 8; ++i) {
        const int h = i + 8 * sel;
        scores[((size_t)bt * HEADS + h) * SEQ + j] =
            acc[i] + kpb[(size_t)(POFF + j) * HEADS + h];
    }
}

// ---------------------------------------------------------------------------
// K3 (streams `key` once, register double-buffered):
//   scores[b][h][j] = (key[j,b,:] . Afold[b,:,h] + bd) / 8
__global__ void k3_ac(const float* __restrict__ key,
                      const _Float16* __restrict__ afold_sw,
                      float* __restrict__ scores) {
    const int lane = threadIdx.x & 31;
    const int j0 = blockIdx.x * 16, bt = blockIdx.y;
    const float* keyb = key + (size_t)j0 * (BATCH * DMODEL) + (size_t)bt * DMODEL;
    const _Float16* bsw = afold_sw + (size_t)bt * 32 * 32 * 16;

    v16h a = frag_a_f32(keyb, BATCH * DMODEL, lane);
    v16h b = *(const v16h*)(bsw + (size_t)lane * 16);
    v8f acc = {};
#pragma unroll 4
    for (int kk = 0; kk < DMODEL / 32 - 1; ++kk) {
        v16h a2 = frag_a_f32(keyb + (kk + 1) * 32, BATCH * DMODEL, lane);
        v16h b2 = *(const v16h*)(bsw + ((size_t)(kk + 1) * 32 + lane) * 16);
        acc = wmma_f16(a, b, acc);
        a = a2; b = b2;
    }
    acc = wmma_f16(a, b, acc);

    const int n = lane & 15, sel = lane >> 4;
#pragma unroll
    for (int i = 0; i < 8; ++i) {
        const size_t idx = ((size_t)bt * HEADS + n) * SEQ + j0 + i + 8 * sel;
        scores[idx] = (acc[i] + scores[idx]) * 0.125f;
    }
}

// ---------------------------------------------------------------------------
// K4: softmax over j per (b,h) row; attn f16 (b,h,j) contiguous in j
__global__ void k4_softmax(const float* __restrict__ scores,
                           _Float16* __restrict__ attn) {
    __shared__ float red[256];
    const int row = blockIdx.x;            // b*16 + h
    const int t = threadIdx.x;
    const float* s = scores + (size_t)row * SEQ;

    float m = -1e30f;
#pragma unroll
    for (int u = 0; u < SEQ / 256; ++u) m = fmaxf(m, s[t + 256 * u]);
    red[t] = m;
    __syncthreads();
    for (int off = 128; off > 0; off >>= 1) {
        if (t < off) red[t] = fmaxf(red[t], red[t + off]);
        __syncthreads();
    }
    m = red[0];
    __syncthreads();

    float ev[SEQ / 256];
    float sum = 0.f;
#pragma unroll
    for (int u = 0; u < SEQ / 256; ++u) {
        ev[u] = __expf(s[t + 256 * u] - m);
        sum += ev[u];
    }
    red[t] = sum;
    __syncthreads();
    for (int off = 128; off > 0; off >>= 1) {
        if (t < off) red[t] += red[t + off];
        __syncthreads();
    }
    const float inv = 1.0f / red[0];
#pragma unroll
    for (int u = 0; u < SEQ / 256; ++u)
        attn[(size_t)row * SEQ + t + 256 * u] = (_Float16)(ev[u] * inv);
}

// ---------------------------------------------------------------------------
// K5 (streams `value` once via async-to-LDS, double buffered):
//   y[b][h][c] = sum_j attn[b,h,j] * value[j,b,c]
// block = 128 threads (4 waves), each wave owns one 16-wide c-tile of a 64-wide
// c-block; the block async-copies a 32j x 64c f32 tile per k-step into LDS.
__global__ void k5_y(const _Float16* __restrict__ attn,
                     const float* __restrict__ value,
                     _Float16* __restrict__ yv) {
    __shared__ __align__(16) float stage[2][32][64];   // 16 KB
    const int tid = threadIdx.x;          // 0..127
    const int lane = tid & 31, wave = tid >> 5;
    const int cblk = blockIdx.x * 64;     // 16 c-blocks
    const int bt = blockIdx.y;
    const _Float16* ab = attn + (size_t)bt * HEADS * SEQ;

    auto issue = [&](int kk, int buf) {
        const int j0 = kk * 32;
#pragma unroll
        for (int s = 0; s < 4; ++s) {
            const int linear = s * 128 + tid;
            const int row = linear >> 4;      // 0..31
            const int seg = linear & 15;      // 0..15 (16B segments)
            const float* gp = value + (size_t)(j0 + row) * (BATCH * DMODEL) +
                              (size_t)bt * DMODEL + cblk + seg * 4;
            const unsigned la = (unsigned)(uintptr_t)&stage[buf][row][seg * 4];
            asm volatile("global_load_async_to_lds_b128 %0, %1, off"
                         :: "v"(la), "v"(gp) : "memory");
        }
    };

    issue(0, 0);
    v8f acc = {};
    const int n = lane & 15, sel = lane >> 4;
    for (int kk = 0; kk < SEQ / 32; ++kk) {
        const int buf = kk & 1;
        if (kk + 1 < SEQ / 32) {
            issue(kk + 1, buf ^ 1);
            asm volatile("s_wait_asynccnt 0x4" ::: "memory");
        } else {
            asm volatile("s_wait_asynccnt 0x0" ::: "memory");
        }
        __syncthreads();
        // A fragment: attn rows h (contiguous j)
        v16h a = frag_a_f16(ab + kk * 32, SEQ, lane);
        // B fragment from LDS: element (k=j, n=c) = stage[buf][k][wave*16 + n]
        float t[16];
#pragma unroll
        for (int e = 0; e < 16; ++e) t[e] = stage[buf][16 * sel + e][wave * 16 + n];
        v16h b;
#pragma unroll
        for (int e = 0; e < 16; ++e) b[e] = (_Float16)t[e];
        acc = wmma_f16(a, b, acc);
        __syncthreads();
    }
#pragma unroll
    for (int i = 0; i < 8; ++i)
        yv[((size_t)bt * HEADS + i + 8 * sel) * DMODEL + cblk + wave * 16 + n] =
            (_Float16)acc[i];
}

// ---------------------------------------------------------------------------
// K6: x[b, h*64+d] = y[b,h,:] @ w_v[:, h*64+d]   (B pre-swizzled)
__global__ void k6_x(const _Float16* __restrict__ yv,
                     const _Float16* __restrict__ wv_sw,
                     _Float16* __restrict__ xf) {
    const int lane = threadIdx.x & 31;
    const int d0 = blockIdx.x * 16, b0 = blockIdx.y * 16, h = blockIdx.z;
    const int nt = h * 4 + blockIdx.x;    // global column tile (h*64+d0)/16
    const _Float16* ybase = yv + ((size_t)b0 * HEADS + h) * DMODEL;

    v16h a = frag_a_f16(ybase, HEADS * DMODEL, lane);
    v16h b = *bsw_tile(wv_sw, 0, nt, lane);
    v8f acc = {};
#pragma unroll 4
    for (int kk = 0; kk < DMODEL / 32 - 1; ++kk) {
        v16h a2 = frag_a_f16(ybase + (kk + 1) * 32, HEADS * DMODEL, lane);
        v16h b2 = *bsw_tile(wv_sw, kk + 1, nt, lane);
        acc = wmma_f16(a, b, acc);
        a = a2; b = b2;
    }
    acc = wmma_f16(a, b, acc);

    const int n = lane & 15, sel = lane >> 4;
#pragma unroll
    for (int i = 0; i < 8; ++i)
        xf[(size_t)(b0 + i + 8 * sel) * DMODEL + h * DK + d0 + n] = (_Float16)acc[i];
}

// ---------------------------------------------------------------------------
// K7: out = x @ w_out + b_out  (64x1024 @ 1024x1024) -> f32  (B pre-swizzled)
__global__ void k7_out(const _Float16* __restrict__ xf,
                       const _Float16* __restrict__ wo_sw,
                       const float* __restrict__ b_out,
                       float* __restrict__ out) {
    const int lane = threadIdx.x & 31;
    const int nt = blockIdx.x, b0 = blockIdx.y * 16;

    v16h a = frag_a_f16(xf + (size_t)b0 * DMODEL, DMODEL, lane);
    v16h b = *bsw_tile(wo_sw, 0, nt, lane);
    v8f acc = {};
#pragma unroll 4
    for (int kk = 0; kk < DMODEL / 32 - 1; ++kk) {
        v16h a2 = frag_a_f16(xf + (size_t)b0 * DMODEL + (kk + 1) * 32, DMODEL, lane);
        v16h b2 = *bsw_tile(wo_sw, kk + 1, nt, lane);
        acc = wmma_f16(a, b, acc);
        a = a2; b = b2;
    }
    acc = wmma_f16(a, b, acc);

    const int n = lane & 15, sel = lane >> 4;
#pragma unroll
    for (int i = 0; i < 8; ++i)
        out[(size_t)(b0 + i + 8 * sel) * DMODEL + nt * 16 + n] = acc[i] + b_out[nt * 16 + n];
}

// ---------------------------------------------------------------------------
extern "C" void kernel_launch(void* const* d_in, const int* in_sizes, int n_in,
                              void* d_out, int out_size, void* d_ws, size_t ws_size,
                              hipStream_t stream) {
    (void)in_sizes; (void)n_in; (void)out_size; (void)ws_size;
    const float* query = (const float*)d_in[0];
    const float* key   = (const float*)d_in[1];
    const float* value = (const float*)d_in[2];
    const float* w_q   = (const float*)d_in[3];
    const float* w_k   = (const float*)d_in[4];
    const float* w_v   = (const float*)d_in[5];
    const float* w_out = (const float*)d_in[6];
    const float* b_out = (const float*)d_in[7];
    const float* kpe   = (const float*)d_in[8];   // (4096, 64)
    const float* kpb   = (const float*)d_in[9];   // (4096, 16)
    const float* qpb   = (const float*)d_in[10];  // (16, 64)
    float* out = (float*)d_out;

    // workspace carve-up (~22.4 MB total)
    char* ws = (char*)d_ws;
    size_t off = 0;
    float*    q_f32    = (float*)(ws + off);    off += (size_t)BATCH * DMODEL * 4;          // 256 KB
    _Float16* afold_sw = (_Float16*)(ws + off); off += (size_t)BATCH * DMODEL * HEADS * 2;  // 2 MB
    float*    scores   = (float*)(ws + off);    off += (size_t)BATCH * HEADS * SEQ * 4;     // 8 MB
    _Float16* attn     = (_Float16*)(ws + off); off += (size_t)BATCH * HEADS * SEQ * 2;     // 4 MB
    _Float16* yv       = (_Float16*)(ws + off); off += (size_t)BATCH * HEADS * DMODEL * 2;  // 2 MB
    _Float16* xf       = (_Float16*)(ws + off); off += (size_t)BATCH * DMODEL * 2;          // 128 KB
    _Float16* wq_sw    = (_Float16*)(ws + off); off += (size_t)DMODEL * DMODEL * 2;         // 2 MB
    _Float16* wv_sw    = (_Float16*)(ws + off); off += (size_t)DMODEL * DMODEL * 2;         // 2 MB
    _Float16* wo_sw    = (_Float16*)(ws + off); off += (size_t)DMODEL * DMODEL * 2;         // 2 MB

    // weight repacks (independent, launch first)
    krepack<<<dim3(32, 16), 256, 0, stream>>>(w_q, wq_sw);
    krepack<<<dim3(32, 16), 256, 0, stream>>>(w_v, wv_sw);
    krepack<<<dim3(32, 16), 256, 0, stream>>>(w_out, wo_sw);

    k1_qproj<<<dim3(DMODEL / 16, BATCH / 16), 32, 0, stream>>>(query, wq_sw, q_f32);
    k2a_afold<<<dim3(DMODEL / 16, BATCH / 16, HEADS), 32, 0, stream>>>(q_f32, w_k, qpb, afold_sw);
    k2b_bd<<<dim3(SEQ / 16, BATCH), 32, 0, stream>>>(q_f32, kpe, kpb, scores);
    k3_ac<<<dim3(SEQ / 16, BATCH), 32, 0, stream>>>(key, afold_sw, scores);
    k4_softmax<<<dim3(BATCH * HEADS), 256, 0, stream>>>(scores, attn);
    k5_y<<<dim3(DMODEL / 64, BATCH), 128, 0, stream>>>(attn, value, yv);
    k6_x<<<dim3(DK / 16, BATCH / 16, HEADS), 32, 0, stream>>>(yv, wv_sw, xf);
    k7_out<<<dim3(DMODEL / 16, BATCH / 16), 32, 0, stream>>>(xf, wo_sw, b_out, out);
}
